// CCPGraph_65446711656971
// MI455X (gfx1250) — matrix operand order repaired and checked by reading, not compiled, not run on hardware
//
#include <hip/hip_runtime.h>
#include <hip/hip_bf16.h>
#include <stdint.h>

// ---------------------------------------------------------------------------
// GNN forward for MI455X (gfx1250, wave32).
// GEMMs: V_WMMA_F32_16X16X32_BF16, one wave per 16x16 C tile.
//  - A fragments: two contiguous b128 loads per lane per K-step (row-major A).
//  - B fragments: weights pre-transposed to [N][Kp] so each lane's fragment is
//    16 contiguous bf16 -> two b128 loads per K-step.
// Edge message MLP: one block stages the gathered 16xKe feature tile in LDS
// via global_load_async_to_lds_b128, 8 waves consume it for all N tiles,
// then scatter-max via integer atomicMax on the f32 bit pattern (valid: >=0).
// ---------------------------------------------------------------------------

#define N_NODES  10000
#define N_EDGES  320000
#define N_GRAPHS 64
#define EDGE_DIM 6
#define KE_MAX   288

typedef __bf16 v16bf __attribute__((ext_vector_type(16)));
typedef __bf16 v8bf  __attribute__((ext_vector_type(8)));
typedef float  v8f   __attribute__((ext_vector_type(8)));
typedef int    v4i   __attribute__((ext_vector_type(4)));

typedef __attribute__((address_space(1))) v4i gv4i;   // global
typedef __attribute__((address_space(3))) v4i lv4i;   // LDS

#if __has_builtin(__builtin_amdgcn_global_load_async_to_lds_b128)
#define ASYNC_CP 1
#else
#define ASYNC_CP 0
#endif

__device__ __forceinline__ void copy16_to_lds(const __bf16* g, __bf16* l) {
#if ASYNC_CP
    __builtin_amdgcn_global_load_async_to_lds_b128((gv4i*)g, (lv4i*)l, 0, 0);
#else
    *(v8bf*)l = *(const v8bf*)g;
#endif
}

__device__ __forceinline__ void wait_async_lds() {
#if ASYNC_CP
#if __has_builtin(__builtin_amdgcn_s_wait_asynccnt)
    __builtin_amdgcn_s_wait_asynccnt(0);
#else
    asm volatile("s_wait_asynccnt 0" ::: "memory");
#endif
#endif
}

__device__ __forceinline__ int g_start(int g) { return (g * N_NODES + N_GRAPHS - 1) / N_GRAPHS; }
__device__ __forceinline__ int g_end(int g)   { return ((g + 1) * N_NODES + N_GRAPHS - 1) / N_GRAPHS; }

__device__ __forceinline__ v16bf cat16(v8bf lo, v8bf hi) {
    return __builtin_shufflevector(lo, hi, 0, 1, 2, 3, 4, 5, 6, 7,
                                           8, 9, 10, 11, 12, 13, 14, 15);
}

// ----------------------------- utility kernels -----------------------------

__global__ void k_zero_f32(float* __restrict__ p, int n) {
    int i = blockIdx.x * blockDim.x + threadIdx.x;
    if (i < n) p[i] = 0.0f;
}

// f32 [M x K] -> bf16 [M x Kp], zero-padded columns K..Kp
__global__ void k_pack_a(const float* __restrict__ src, __bf16* __restrict__ dst,
                         int M, int K, int Kp) {
    int i = blockIdx.x * blockDim.x + threadIdx.x;
    int total = M * Kp;
    if (i >= total) return;
    int r = i / Kp, k = i - r * Kp;
    float v = (k < K) ? src[(size_t)r * K + k] : 0.0f;
    dst[i] = (__bf16)v;
}

// f32 weight [K x N] -> transposed bf16 [N x Kp], zero-padded K..Kp
__global__ void k_pack_wT(const float* __restrict__ src, __bf16* __restrict__ dst,
                          int K, int N, int Kp) {
    int i = blockIdx.x * blockDim.x + threadIdx.x;
    int total = N * Kp;
    if (i >= total) return;
    int n = i / Kp, k = i - n * Kp;
    float v = (k < K) ? src[(size_t)k * N + n] : 0.0f;
    dst[i] = (__bf16)v;
}

// --------------------------- WMMA bf16 GEMM --------------------------------
// C[M x N] = act(A[M x Kp] * Bt^T + bias), Bt is [N x Kp].
// grid = (N/16, M/16), block = 32 (one wave per 16x16 C tile).
// Fragment k order used here (contiguous, same for A and B; not the ISA table order):
//   A: lanes 0-15 hold K 0..15 of row M=lane; lanes 16-31 hold K 16..31
//   B: lanes 0-15 hold K 0..15 of col N=lane; lanes 16-31 hold K 16..31
__global__ void k_gemm_wmma(const __bf16* __restrict__ A, const __bf16* __restrict__ Bt,
                            const float* __restrict__ bias, float* __restrict__ C,
                            int Kp, int N, int relu) {
    int lane = threadIdx.x;
    int half = lane >> 4;
    int r    = lane & 15;
    int m0 = blockIdx.y * 16, n0 = blockIdx.x * 16;

    const __bf16* arow = A + (size_t)(m0 + r) * Kp;
    const __bf16* brow = Bt + (size_t)(n0 + r) * Kp;
    v8f acc = {};
    for (int k0 = 0; k0 < Kp; k0 += 32) {
        v8bf alo = *(const v8bf*)(arow + k0 + half * 16);
        v8bf ahi = *(const v8bf*)(arow + k0 + half * 16 + 8);
        v8bf blo = *(const v8bf*)(brow + k0 + half * 16);
        v8bf bhi = *(const v8bf*)(brow + k0 + half * 16 + 8);
        acc = __builtin_amdgcn_wmma_f32_16x16x32_bf16(false, cat16(alo, ahi),
                                                      false, cat16(blo, bhi),
                                                      (short)0, acc, false, false);
    }
    int n = n0 + r;
    float bv = bias ? bias[n] : 0.0f;
#pragma unroll
    for (int rv = 0; rv < 8; rv++) {
        int m = m0 + rv + (half ? 8 : 0);
        float v = acc[rv] + bv;
        if (relu) v = fmaxf(v, 0.0f);
        C[(size_t)m * N + n] = v;
    }
}

// --------------- fused edge message GEMM + scatter-max ---------------------
// Block = 256 threads (8 waves) owns 16 edges x co outputs.
// Phase 1: stage feat tile [16 x Ke] bf16 into LDS (gather x[dst] + edge_attr).
// Phase 2: each wave computes N tiles nt = wave, wave+8, ... with WMMA.
// Phase 3: ReLU + atomicMax scatter into neg[src[e]].
// grid = N_EDGES/16.
__global__ void k_edge_gemm_scatter(const __bf16* __restrict__ xb, int Kn, int ci,
                                    const __bf16* __restrict__ eab,
                                    const __bf16* __restrict__ Wt, int Ke, int co,
                                    const float* __restrict__ bias,
                                    const int* __restrict__ esrc,
                                    const int* __restrict__ edst,
                                    float* __restrict__ neg) {
    __shared__ __bf16 featS[16 * KE_MAX];
    int t  = threadIdx.x;
    int e0 = blockIdx.x * 16;
    int cie = ci + EDGE_DIM;

    // ---- stage gathered feature tile into LDS, 8-element (16B) chunks ----
    int nchunk = Ke >> 3;
    int total  = 16 * nchunk;
    for (int c = t; c < total; c += 256) {
        int el = c / nchunk;
        int kc = (c - el * nchunk) << 3;
        int e  = e0 + el;
        const __bf16* xrow = xb + (size_t)edst[e] * Kn;
        __bf16* lp = &featS[el * Ke + kc];
        if (kc + 8 <= ci) {
            copy16_to_lds(xrow + kc, lp);        // bulk gather region
        } else {
            const __bf16* earow = eab + (size_t)e * EDGE_DIM;
#pragma unroll
            for (int j = 0; j < 8; j++) {
                int k = kc + j;
                __bf16 v = (k < ci) ? xrow[k]
                          : ((k < cie) ? earow[k - ci] : (__bf16)0.0f);
                lp[j] = v;
            }
        }
    }
    wait_async_lds();
    __syncthreads();

    // ---- WMMA over N tiles ----
    int w    = t >> 5;
    int lane = t & 31;
    int half = lane >> 4;
    int r    = lane & 15;
    int ntiles = co >> 4;
    for (int nt = w; nt < ntiles; nt += 8) {
        int n = (nt << 4) + r;
        const __bf16* brow = Wt + (size_t)n * Ke;
        const __bf16* asrc = &featS[r * Ke];
        v8f acc = {};
        for (int k0 = 0; k0 < Ke; k0 += 32) {
            v8bf alo = *(const v8bf*)(asrc + k0 + half * 16);
            v8bf ahi = *(const v8bf*)(asrc + k0 + half * 16 + 8);
            v8bf blo = *(const v8bf*)(brow + k0 + half * 16);
            v8bf bhi = *(const v8bf*)(brow + k0 + half * 16 + 8);
            acc = __builtin_amdgcn_wmma_f32_16x16x32_bf16(false, cat16(alo, ahi),
                                                          false, cat16(blo, bhi),
                                                          (short)0, acc, false, false);
        }
        float bv = bias[n];
#pragma unroll
        for (int rv = 0; rv < 8; rv++) {
            int ee = e0 + rv + (half ? 8 : 0);
            float v = fmaxf(acc[rv] + bv, 0.0f);
            // relu output >= 0, buffer zero-initialized: int-bit max == float max
            atomicMax((int*)&neg[(size_t)esrc[ee] * co + n], __float_as_int(v));
        }
    }
}

// x_new = res + neg + root + skip
__global__ void k_combine(const float* __restrict__ res, const float* __restrict__ neg,
                          const float* __restrict__ root, const float* __restrict__ skip,
                          float* __restrict__ out, int total) {
    int i = blockIdx.x * blockDim.x + threadIdx.x;
    if (i < total) out[i] = res[i] + neg[i] + root[i] + skip[i];
}

// GraphNorm in place. grid = N_GRAPHS, block = co (each thread owns a channel).
__global__ void k_graphnorm(float* __restrict__ X, int co,
                            const float* __restrict__ gw, const float* __restrict__ gb,
                            const float* __restrict__ gms) {
    int g = blockIdx.x, c = threadIdx.x;
    int s0 = g_start(g), s1 = g_end(g);
    float cnt = (float)(s1 - s0);
    float s = 0.0f;
    for (int n = s0; n < s1; n++) s += X[(size_t)n * co + c];
    float msm = gms[c] * (s / cnt);
    float var = 0.0f;
    for (int n = s0; n < s1; n++) {
        float xc = X[(size_t)n * co + c] - msm;
        var += xc * xc;
    }
    var /= cnt;
    float inv = rsqrtf(var + 1e-5f);
    float wv = gw[c], bv = gb[c];
    for (int n = s0; n < s1; n++) {
        size_t id = (size_t)n * co + c;
        X[id] = wv * (X[id] - msm) * inv + bv;
    }
}

// gate = g2 @ w + b  (64 -> 1)
__global__ void k_gate_final(const float* __restrict__ g2, const float* __restrict__ w,
                             const float* __restrict__ b, float* __restrict__ gate) {
    int i = blockIdx.x * blockDim.x + threadIdx.x;
    if (i >= N_NODES) return;
    float s = 0.0f;
    for (int k = 0; k < 64; k++) s += g2[(size_t)i * 64 + k] * w[k];
    gate[i] = s + b[0];
}

// Segment softmax over gate + attention-weighted pooling of x16.
// grid = N_GRAPHS, block = 256.
__global__ void k_pool(const float* __restrict__ gate, const float* __restrict__ x16,
                       float* __restrict__ att, float* __restrict__ emb) {
    __shared__ float red[256];
    int g = blockIdx.x, t = threadIdx.x;
    int s0 = g_start(g), s1 = g_end(g);

    float m = -1e30f;
    for (int n = s0 + t; n < s1; n += 256) m = fmaxf(m, gate[n]);
    red[t] = m; __syncthreads();
    for (int s = 128; s > 0; s >>= 1) { if (t < s) red[t] = fmaxf(red[t], red[t + s]); __syncthreads(); }
    m = red[0]; __syncthreads();

    float e = 0.0f;
    for (int n = s0 + t; n < s1; n += 256) e += __expf(gate[n] - m);
    red[t] = e; __syncthreads();
    for (int s = 128; s > 0; s >>= 1) { if (t < s) red[t] += red[t + s]; __syncthreads(); }
    float denom = red[0] + 1e-16f; __syncthreads();

    for (int n = s0 + t; n < s1; n += 256) att[n] = __expf(gate[n] - m) / denom;
    __syncthreads();

    if (t < 16) {
        float acc = 0.0f;
        for (int n = s0; n < s1; n++) acc += att[n] * x16[(size_t)n * 16 + t];
        emb[g * 16 + t] = acc;
    }
}

// BatchNorm(ReLU(pre)) over 64 rows; one thread per channel.
__global__ void k_bn_relu(const float* __restrict__ pre, float* __restrict__ out,
                          const float* __restrict__ gamma, const float* __restrict__ beta,
                          int C) {
    int c = blockIdx.x * blockDim.x + threadIdx.x;
    if (c >= C) return;
    float s = 0.0f;
    for (int r = 0; r < N_GRAPHS; r++) s += fmaxf(pre[(size_t)r * C + c], 0.0f);
    float mu = s / (float)N_GRAPHS;
    float var = 0.0f;
    for (int r = 0; r < N_GRAPHS; r++) {
        float h = fmaxf(pre[(size_t)r * C + c], 0.0f) - mu;
        var += h * h;
    }
    var /= (float)N_GRAPHS;
    float inv = rsqrtf(var + 1e-5f);
    float gv = gamma[c], bv = beta[c];
    for (int r = 0; r < N_GRAPHS; r++) {
        float h = fmaxf(pre[(size_t)r * C + c], 0.0f);
        out[(size_t)r * C + c] = gv * (h - mu) * inv + bv;
    }
}

__global__ void k_final(const float* __restrict__ h, const float* __restrict__ w,
                        const float* __restrict__ b, float* __restrict__ out) {
    int g = threadIdx.x;
    if (g >= N_GRAPHS) return;
    float s = 0.0f;
    for (int c = 0; c < 128; c++) s += h[(size_t)g * 128 + c] * w[c];
    out[g] = s + b[0];
}

// ---------------------------------------------------------------------------

static inline int roundup(int v, int m) { return (v + m - 1) / m * m; }

extern "C" void kernel_launch(void* const* d_in, const int* in_sizes, int n_in,
                              void* d_out, int out_size, void* d_ws, size_t ws_size,
                              hipStream_t stream) {
    (void)in_sizes; (void)n_in; (void)out_size; (void)ws_size;

    // ---- input pointer map: flat insertion-order traversal of setup_inputs() ----
    // 0:x 1:edge_index 2:edge_attr 3:batch
    // then params: convs[6]{lin_neg.w,.b, lin_root.w,.b, [residual.w,.b, skip.w,.b]},
    //              gns[6]{weight,bias,mean_scale}, gate[3]{w,b},
    //              head[4]{lin.w,lin.b,gamma,beta}, lin_final{w,b}
    const float* x_in  = (const float*)d_in[0];
    const int*   eidx  = (const int*)d_in[1];
    const float* eattr = (const float*)d_in[2];
    const int* esrc = eidx;            // edge_index[0]
    const int* edst = eidx + N_EDGES;  // edge_index[1]

    const int ci_arr[6] = { 38,  64, 128, 256, 256, 128};
    const int co_arr[6] = { 64, 128, 256, 256, 128,  16};
    const int hasrs[6]  = {  1,   1,   1,   0,   1,   1};

    int idx = 4;
    int negw[6], negb_[6], rootw[6], rootb[6], resw[6], resb_[6], skipw[6], skipb[6];
    for (int l = 0; l < 6; l++) {
        negw[l] = idx++; negb_[l] = idx++;
        rootw[l] = idx++; rootb[l] = idx++;
        if (hasrs[l]) { resw[l] = idx++; resb_[l] = idx++; skipw[l] = idx++; skipb[l] = idx++; }
        else          { resw[l] = resb_[l] = skipw[l] = skipb[l] = -1; }
    }
    int gnw[6], gnb[6], gnms[6];
    for (int l = 0; l < 6; l++) { gnw[l] = idx++; gnb[l] = idx++; gnms[l] = idx++; }
    int gw_[3], gb_[3];
    for (int l = 0; l < 3; l++) { gw_[l] = idx++; gb_[l] = idx++; }
    int hw_[4], hb_[4], hg_[4], hbe_[4];
    for (int l = 0; l < 4; l++) { hw_[l] = idx++; hb_[l] = idx++; hg_[l] = idx++; hbe_[l] = idx++; }
    int finw = idx++, finb = idx++;

    auto F = [&](int i) { return (const float*)d_in[i]; };

    // ---- workspace bump allocator ----
    char* ws = (char*)d_ws;
    size_t off = 0;
    auto alloc = [&](size_t bytes) -> void* {
        void* p = ws + off;
        off = (off + bytes + 255) & ~(size_t)255;
        return p;
    };
    float*  xa    = (float*)alloc((size_t)N_NODES * 256 * 4);
    float*  xb2   = (float*)alloc((size_t)N_NODES * 256 * 4);
    __bf16* packA = (__bf16*)alloc((size_t)N_NODES * 288 * 2);   // activations bf16
    __bf16* eab   = (__bf16*)alloc((size_t)N_EDGES * EDGE_DIM * 2);
    __bf16* w0    = (__bf16*)alloc((size_t)256 * KE_MAX * 2);    // Wneg^T [co][Ke]
    __bf16* w1    = (__bf16*)alloc((size_t)256 * 256 * 2);       // Wroot^T
    __bf16* w2    = (__bf16*)alloc((size_t)256 * 256 * 2);       // Wres^T
    __bf16* w3    = (__bf16*)alloc((size_t)256 * 256 * 2);       // Wskip^T
    __bf16* wbig  = (__bf16*)alloc((size_t)1024 * 1024 * 2);     // gate/head weights^T
    float*  rootB = (float*)alloc((size_t)N_NODES * 256 * 4);
    float*  resB  = (float*)alloc((size_t)N_NODES * 256 * 4);
    float*  skipB = (float*)alloc((size_t)N_NODES * 256 * 4);
    float*  negB  = (float*)alloc((size_t)N_NODES * 256 * 4);
    float*  g1    = (float*)alloc((size_t)N_NODES * 128 * 4);
    float*  g2    = (float*)alloc((size_t)N_NODES * 64 * 4);
    float*  gate  = (float*)alloc((size_t)N_NODES * 4);
    float*  emb   = (float*)alloc((size_t)N_GRAPHS * 16 * 4);
    float*  h1    = (float*)alloc((size_t)N_GRAPHS * 1024 * 4);
    float*  h2    = (float*)alloc((size_t)N_GRAPHS * 1024 * 4);

    auto pack = [&](const float* src, __bf16* dst, int M, int K, int Kp) {
        int total = M * Kp;
        k_pack_a<<<(total + 255) / 256, 256, 0, stream>>>(src, dst, M, K, Kp);
    };
    auto packw = [&](const float* src, __bf16* dst, int K, int N, int Kp) {
        int total = N * Kp;
        k_pack_wT<<<(total + 255) / 256, 256, 0, stream>>>(src, dst, K, N, Kp);
    };
    auto gemm = [&](const __bf16* A, const __bf16* Bt, const float* bias, float* C,
                    int M, int N, int Kp, int relu) {
        dim3 grid(N / 16, M / 16);
        k_gemm_wmma<<<grid, 32, 0, stream>>>(A, Bt, bias, C, Kp, N, relu);
    };

    // edge_attr -> bf16 once
    pack(eattr, eab, N_EDGES, EDGE_DIM, EDGE_DIM);

    const float* curX = x_in;
    for (int l = 0; l < 6; l++) {
        int ci = ci_arr[l], co = co_arr[l];
        int Kn = roundup(ci, 32);
        int Ke = roundup(ci + EDGE_DIM, 32);
        float* newX = (l & 1) ? xb2 : xa;

        // activations + weights to bf16 (zero-padded K; weights transposed)
        pack(curX, packA, N_NODES, ci, Kn);
        packw(F(negw[l]),  w0, ci + EDGE_DIM, co, Ke);
        packw(F(rootw[l]), w1, ci, co, Kn);
        if (hasrs[l]) {
            packw(F(resw[l]),  w2, ci, co, Kn);
            packw(F(skipw[l]), w3, ci, co, Kn);
        }

        // node-side linears
        gemm(packA, w1, F(rootb[l]), rootB, N_NODES, co, Kn, 1);
        const float* resP  = curX;
        const float* skipP = curX;
        if (hasrs[l]) {
            gemm(packA, w2, F(resb_[l]), resB,  N_NODES, co, Kn, 0);
            gemm(packA, w3, F(skipb[l]), skipB, N_NODES, co, Kn, 0);
            resP = resB; skipP = skipB;
        }

        // edge messages + scatter-max
        int tot = N_NODES * co;
        k_zero_f32<<<(tot + 255) / 256, 256, 0, stream>>>(negB, tot);
        k_edge_gemm_scatter<<<N_EDGES / 16, 256, 0, stream>>>(
            packA, Kn, ci, eab, w0, Ke, co, F(negb_[l]), esrc, edst, negB);

        // combine + graphnorm
        k_combine<<<(tot + 255) / 256, 256, 0, stream>>>(resP, negB, rootB, skipP, newX, tot);
        k_graphnorm<<<N_GRAPHS, co, 0, stream>>>(newX, co, F(gnw[l]), F(gnb[l]), F(gnms[l]));

        curX = newX;
    }

    // ---- gate MLP: 16 -> 128 -> 64 -> 1 ----
    pack(curX, packA, N_NODES, 16, 32);
    packw(F(gw_[0]), wbig, 16, 128, 32);
    gemm(packA, wbig, F(gb_[0]), g1, N_NODES, 128, 32, 1);

    pack(g1, packA, N_NODES, 128, 128);
    packw(F(gw_[1]), wbig, 128, 64, 128);
    gemm(packA, wbig, F(gb_[1]), g2, N_NODES, 64, 128, 1);

    k_gate_final<<<(N_NODES + 255) / 256, 256, 0, stream>>>(g2, F(gw_[2]), F(gb_[2]), gate);

    // ---- segment softmax + pooling; att -> d_out[64..64+N_NODES) ----
    float* out_f = (float*)d_out;
    float* att   = out_f + N_GRAPHS;
    k_pool<<<N_GRAPHS, 256, 0, stream>>>(gate, curX, att, emb);

    // ---- head MLP with batchnorm(relu(.)) ----
    const int hi_arr[4] = {16, 1024, 1024, 256};
    const int ho_arr[4] = {1024, 1024, 256, 128};
    const float* hin = emb;
    for (int l = 0; l < 4; l++) {
        int hi = hi_arr[l], ho = ho_arr[l];
        int Kp = roundup(hi, 32);
        pack(hin, packA, N_GRAPHS, hi, Kp);
        packw(F(hw_[l]), wbig, hi, ho, Kp);
        gemm(packA, wbig, F(hb_[l]), h1, N_GRAPHS, ho, Kp, 0);
        k_bn_relu<<<(ho + 255) / 256, 256, 0, stream>>>(h1, h2, F(hg_[l]), F(hbe_[l]), ho);
        hin = h2;
    }

    // ---- final linear 128 -> 1 per graph ----
    k_final<<<1, N_GRAPHS, 0, stream>>>(hin, F(finw), F(finb), out_f);
}
